// AttentionPerLabelWordLevel_41472204210408
// MI455X (gfx1250) — matrix operand order repaired and checked by reading, not compiled, not run on hardware
//
#include <hip/hip_runtime.h>
#include <hip/hip_bf16.h>

// CDNA5 / gfx1250, wave32.  All matrix math via v_wmma_f32_16x16x32_bf16.

typedef __attribute__((ext_vector_type(16))) __bf16 v16bf;
typedef __attribute__((ext_vector_type(8)))  float  v8f;

#define HID     512
#define NSENT   100
#define SLEN    25
#define NCLS    50
#define NB      32
#define TOTROWS (NB * NSENT * SLEN)   // 80000
#define NSENTTOT (NB * NSENT)         // 3200

static __device__ __forceinline__ v8f wmma_bf16(v16bf a, v16bf b, v8f c) {
  // 8 args: (neg_a, A, neg_b, B, c_mod, C, reuse_a, reuse_b)
  return __builtin_amdgcn_wmma_f32_16x16x32_bf16(false, a, false, b, (short)0, c,
                                                 false, false);
}

// A-matrix (16x32 bf16) per-lane K index for VGPR dword v (2 bf16 per dword)
static __device__ __forceinline__ int a_kidx(int v, int hi) {
  return ((v < 4) ? 2 * v : 2 * v + 8) + hi * 8;
}
// B-matrix (32x16 bf16) per-lane K index
static __device__ __forceinline__ int b_kidx(int v, int hi) {
  return ((v < 4) ? 2 * v : 2 * v + 8) + hi * 8;
}

// ---------------------------------------------------------------------------
// Kernel 1:  h[t,o] = tanh( sum_i X[t,i] * W[o,i] + b[o] ), stored as bf16.
// One wave computes a 32x32 tile: 2x2 WMMA accumulators, K stepped by 32.
// 80000/32 = 2500 row tiles x 512/32 = 16 col tiles = 40000 waves.
// ---------------------------------------------------------------------------
__global__ __launch_bounds__(256) void k1_linear_tanh(
    const float* __restrict__ X, const float* __restrict__ W,
    const float* __restrict__ bias, __bf16* __restrict__ Hout)
{
  const int lane = threadIdx.x & 31;
  const int hi   = lane >> 4;
  const int ll   = lane & 15;
  const int wave = blockIdx.x * (blockDim.x >> 5) + (threadIdx.x >> 5);
  const int rt   = wave >> 4;          // 0..2499
  const int ct   = wave & 15;          // 0..15
  const int row0 = rt * 32;
  const int col0 = ct * 32;

  v8f acc[2][2];
#pragma unroll
  for (int i = 0; i < 2; ++i)
#pragma unroll
    for (int j = 0; j < 2; ++j)
      acc[i][j] = (v8f){0.f, 0.f, 0.f, 0.f, 0.f, 0.f, 0.f, 0.f};

  for (int kk = 0; kk < HID; kk += 32) {
    v16bf A[2], Bv[2];
#pragma unroll
    for (int mt = 0; mt < 2; ++mt) {
      const float* xr = X + (size_t)(row0 + mt * 16 + ll) * HID + kk;
#pragma unroll
      for (int v = 0; v < 8; ++v) {
        const int k = a_kidx(v, hi);
        const float2 f = *(const float2*)(xr + k);
        A[mt][2 * v]     = (__bf16)f.x;
        A[mt][2 * v + 1] = (__bf16)f.y;
      }
    }
#pragma unroll
    for (int nt = 0; nt < 2; ++nt) {
      // B[k,n] = W[n,k]; W row n is contiguous in k -> float2 loads
      const float* wr = W + (size_t)(col0 + nt * 16 + ll) * HID + kk;
#pragma unroll
      for (int v = 0; v < 8; ++v) {
        const int k = b_kidx(v, hi);
        const float2 f = *(const float2*)(wr + k);
        Bv[nt][2 * v]     = (__bf16)f.x;
        Bv[nt][2 * v + 1] = (__bf16)f.y;
      }
    }
#pragma unroll
    for (int mt = 0; mt < 2; ++mt)
#pragma unroll
      for (int nt = 0; nt < 2; ++nt)
        acc[mt][nt] = wmma_bf16(A[mt], Bv[nt], acc[mt][nt]);
  }

#pragma unroll
  for (int mt = 0; mt < 2; ++mt) {
#pragma unroll
    for (int nt = 0; nt < 2; ++nt) {
      const int col = col0 + nt * 16 + ll;
      const float bv = bias[col];
#pragma unroll
      for (int r = 0; r < 8; ++r) {
        const int row = row0 + mt * 16 + r + hi * 8;
        Hout[(size_t)row * HID + col] = (__bf16)tanhf(acc[mt][nt][r] + bv);
      }
    }
  }
}

// ---------------------------------------------------------------------------
// Kernel 2: per sentence (one wave each):
//   logits[l,c] = sum_o h[l,o]*ctx[c,o]        (WMMA, M=32pad, N=64pad, K=512)
//   m[c] = max_l logits ; attn = softmax_c(logits - m[c])
//   out[c,o] = sum_l attn[l,c] * x[l,o]        (WMMA, attn^T via LDS, K=32pad)
// ---------------------------------------------------------------------------
__global__ __launch_bounds__(256) void k2_attn_out(
    const float* __restrict__ X, const __bf16* __restrict__ Hs,
    const float* __restrict__ ctx, float* __restrict__ Out)
{
  __shared__ __bf16 attnT[8][64 * 32];   // per-wave [c][l] (bf16), 4 KB each
  const int lane  = threadIdx.x & 31;
  const int hi    = lane >> 4;
  const int ll    = lane & 15;
  const int wslot = threadIdx.x >> 5;
  const int sent  = blockIdx.x * 8 + wslot;   // 0..3199

  // ---- Phase 1: logits = h @ ctx^T ----
  v8f lg[2][4];
#pragma unroll
  for (int i = 0; i < 2; ++i)
#pragma unroll
    for (int j = 0; j < 4; ++j)
      lg[i][j] = (v8f){0.f, 0.f, 0.f, 0.f, 0.f, 0.f, 0.f, 0.f};

  const __bf16* hb = Hs + (size_t)sent * SLEN * HID;
  for (int kk = 0; kk < HID; kk += 32) {
    v16bf A[2];
#pragma unroll
    for (int mt = 0; mt < 2; ++mt) {
      const int l = mt * 16 + ll;
      union { v16bf v; unsigned u[8]; } av;
      if (l < SLEN) {
        const unsigned* hp = (const unsigned*)(hb + (size_t)l * HID + kk);
#pragma unroll
        for (int v = 0; v < 8; ++v) av.u[v] = hp[a_kidx(v, hi) >> 1];
      } else {
#pragma unroll
        for (int v = 0; v < 8; ++v) av.u[v] = 0u;
      }
      A[mt] = av.v;
    }
    v16bf Bv[4];
#pragma unroll
    for (int nt = 0; nt < 4; ++nt) {
      const int c = nt * 16 + ll;
      if (c < NCLS) {
        const float* cr = ctx + (size_t)c * HID + kk;
#pragma unroll
        for (int v = 0; v < 8; ++v) {
          const int k = b_kidx(v, hi);
          const float2 f = *(const float2*)(cr + k);
          Bv[nt][2 * v]     = (__bf16)f.x;
          Bv[nt][2 * v + 1] = (__bf16)f.y;
        }
      } else {
#pragma unroll
        for (int v = 0; v < 16; ++v) Bv[nt][v] = (__bf16)0.f;
      }
    }
#pragma unroll
    for (int mt = 0; mt < 2; ++mt)
#pragma unroll
      for (int nt = 0; nt < 4; ++nt)
        lg[mt][nt] = wmma_bf16(A[mt], Bv[nt], lg[mt][nt]);
  }

  // ---- Phase 2: per-class max over words, softmax over classes ----
  // C/D layout: reg r of lg[mt][nt] is element (l = mt*16 + r + hi*8,
  //                                             c = nt*16 + ll)
  float cmax[4];
#pragma unroll
  for (int nt = 0; nt < 4; ++nt) {
    float mx = -1e30f;
#pragma unroll
    for (int mt = 0; mt < 2; ++mt)
#pragma unroll
      for (int r = 0; r < 8; ++r) {
        const int l = mt * 16 + r + hi * 8;
        if (l < SLEN) mx = fmaxf(mx, lg[mt][nt][r]);
      }
    mx = fmaxf(mx, __shfl_xor(mx, 16, 32));   // partner lane: other 16 rows
    cmax[nt] = mx;
  }
#pragma unroll
  for (int mt = 0; mt < 2; ++mt)
#pragma unroll
    for (int nt = 0; nt < 4; ++nt)
#pragma unroll
      for (int r = 0; r < 8; ++r) lg[mt][nt][r] -= cmax[nt];

  float rmax[2][8], rsum[2][8];
#pragma unroll
  for (int mt = 0; mt < 2; ++mt) {
#pragma unroll
    for (int r = 0; r < 8; ++r) {
      float rm = -1e30f;
#pragma unroll
      for (int nt = 0; nt < 4; ++nt) {
        const int c = nt * 16 + ll;
        if (c < NCLS) rm = fmaxf(rm, lg[mt][nt][r]);
      }
      rm = fmaxf(rm, __shfl_xor(rm, 1, 32));
      rm = fmaxf(rm, __shfl_xor(rm, 2, 32));
      rm = fmaxf(rm, __shfl_xor(rm, 4, 32));
      rm = fmaxf(rm, __shfl_xor(rm, 8, 32));
      rmax[mt][r] = rm;
    }
  }
#pragma unroll
  for (int mt = 0; mt < 2; ++mt) {
#pragma unroll
    for (int r = 0; r < 8; ++r) {
      float s = 0.f;
      const int l = mt * 16 + r + hi * 8;
#pragma unroll
      for (int nt = 0; nt < 4; ++nt) {
        const int c = nt * 16 + ll;
        const float ev = (l < SLEN && c < NCLS)
                             ? __expf(lg[mt][nt][r] - rmax[mt][r]) : 0.f;
        lg[mt][nt][r] = ev;
        s += ev;
      }
      s += __shfl_xor(s, 1, 32);
      s += __shfl_xor(s, 2, 32);
      s += __shfl_xor(s, 4, 32);
      s += __shfl_xor(s, 8, 32);
      rsum[mt][r] = s;
    }
  }

  // normalize + write attn^T (shape [64 c][32 l]) to LDS as bf16
  __bf16* at = &attnT[wslot][0];
#pragma unroll
  for (int mt = 0; mt < 2; ++mt)
#pragma unroll
    for (int nt = 0; nt < 4; ++nt)
#pragma unroll
      for (int r = 0; r < 8; ++r) {
        const int l = mt * 16 + r + hi * 8;
        const int c = nt * 16 + ll;
        const float a = (l < SLEN) ? (lg[mt][nt][r] / rsum[mt][r]) : 0.f;
        at[c * 32 + l] = (__bf16)a;
      }

  // ---- Phase 3: out = attn^T @ x   (M=64pad over C, K=32pad over L, N=512) ----
  v16bf AT[4];
#pragma unroll
  for (int mt2 = 0; mt2 < 4; ++mt2) {
    const unsigned* ap = (const unsigned*)(at + (mt2 * 16 + ll) * 32);
    union { v16bf v; unsigned u[8]; } av;
#pragma unroll
    for (int v = 0; v < 8; ++v) av.u[v] = ap[a_kidx(v, hi) >> 1];
    AT[mt2] = av.v;
  }

  const float* xb = X + (size_t)sent * SLEN * HID;
  float* ob = Out + (size_t)sent * NCLS * HID;
  const v8f z = (v8f){0.f, 0.f, 0.f, 0.f, 0.f, 0.f, 0.f, 0.f};
  for (int nt2 = 0; nt2 < 32; ++nt2) {
    const int o = nt2 * 16 + ll;
    v16bf Bv;
#pragma unroll
    for (int v = 0; v < 8; ++v) {
      const int k = b_kidx(v, hi);   // word index l
      Bv[2 * v]     = (k     < SLEN) ? (__bf16)xb[(size_t)k       * HID + o]
                                     : (__bf16)0.f;
      Bv[2 * v + 1] = (k + 1 < SLEN) ? (__bf16)xb[(size_t)(k + 1) * HID + o]
                                     : (__bf16)0.f;
    }
#pragma unroll
    for (int mt2 = 0; mt2 < 4; ++mt2) {
      const v8f oc = wmma_bf16(AT[mt2], Bv, z);
#pragma unroll
      for (int r = 0; r < 8; ++r) {
        const int c = mt2 * 16 + r + hi * 8;
        if (c < NCLS) ob[(size_t)c * HID + o] = oc[r];
      }
    }
  }
}

// ---------------------------------------------------------------------------
extern "C" void kernel_launch(void* const* d_in, const int* in_sizes, int n_in,
                              void* d_out, int out_size, void* d_ws, size_t ws_size,
                              hipStream_t stream) {
  const float* X    = (const float*)d_in[0];   // [32, 2500, 512]
  const float* W    = (const float*)d_in[1];   // [512, 512]
  const float* bias = (const float*)d_in[2];   // [512]
  const float* ctx  = (const float*)d_in[3];   // [50, 512]
  float* Out = (float*)d_out;                  // [32, 100, 50, 512]
  __bf16* Hws = (__bf16*)d_ws;                 // h in bf16: 80000*512*2 = 81.92 MB

  // K1: 2500 row-tiles x 16 col-tiles = 40000 waves / 8 per block
  k1_linear_tanh<<<5000, 256, 0, stream>>>(X, W, bias, Hws);
  // K2: 3200 sentences, one wave each, 8 per block
  k2_attn_out<<<400, 256, 0, stream>>>(X, Hws, ctx, Out);
}
